// SpatialTransformer_23072564314369
// MI455X (gfx1250) — matrix operand order repaired and checked by reading, not compiled, not run on hardware
//
#include <hip/hip_runtime.h>

typedef __bf16 bhalf;
typedef __bf16 v16bf __attribute__((ext_vector_type(16)));
typedef __bf16 v8bf  __attribute__((ext_vector_type(8)));
typedef float  v8f   __attribute__((ext_vector_type(8)));
typedef unsigned int u32x4 __attribute__((ext_vector_type(4)));
typedef int          i32x4 __attribute__((ext_vector_type(4)));
typedef int          i32x8 __attribute__((ext_vector_type(8)));

#define WMMA_BF16(a, b, c) \
  __builtin_amdgcn_wmma_f32_16x16x32_bf16(false, (a), false, (b), (short)0, (c), false, false)

// ---------------------------------------------------------------------------
// Pack weights: k1 (3*3*3*32 fp32, HWIO flat = K*32+co) -> bf16 [32][32]
// zero-padded K=27..31; k2 (flat = K*64+co) -> bf16. Zero FC1 accumulator.
// ---------------------------------------------------------------------------
__global__ void pack_kernel(const float* __restrict__ k1, const float* __restrict__ k2,
                            bhalf* __restrict__ wk1, bhalf* __restrict__ wk2,
                            float* __restrict__ y) {
  int i = blockIdx.x * blockDim.x + threadIdx.x;
  if (i < 1024)  wk1[i] = (i < 864) ? (bhalf)k1[i] : (bhalf)0.0f;
  if (i < 18432) wk2[i] = (bhalf)k2[i];
  if (i < 4096)  y[i] = 0.0f;
}

// ---------------------------------------------------------------------------
// conv1: implicit GEMM, M = 32*128*128 pixels, N = 32, K = 27 (pad 32).
// One wave = one 16-pixel tile, both 16-channel N-tiles (A reused, 2 WMMAs).
// ---------------------------------------------------------------------------
__global__ void __launch_bounds__(256) conv1_kernel(const float* __restrict__ in,
                                                    const float* __restrict__ b1,
                                                    const bhalf* __restrict__ wk1,
                                                    bhalf* __restrict__ act1) {
  int wid  = blockIdx.x * (blockDim.x >> 5) + (threadIdx.x >> 5); // pixel tile
  int lane = threadIdx.x & 31;
  int row   = lane & 15;
  int kbase = (lane < 16) ? 0 : 8;

  int p   = wid * 16 + row;
  int b   = p >> 14;
  int rem = p & 16383;
  int oy  = rem >> 7, ox = rem & 127;

  v16bf a;
#pragma unroll
  for (int h = 0; h < 16; ++h) {
    int K = kbase + ((h < 8) ? h : (h + 8));   // {kb..kb+7, kb+16..kb+23}
    float v = 0.0f;
    if (K < 27) {
      int ci = K % 3;
      int t  = K / 3;
      int ky = t / 3, kx = t % 3;
      int iy = 2 * oy + ky - 1;
      int ix = 2 * ox + kx - 1;
      if (iy >= 0 && iy < 256 && ix >= 0 && ix < 256)
        v = in[((b * 256 + iy) * 256 + ix) * 3 + ci];
    }
    a[h] = (bhalf)v;
  }

  v8f acc[2] = {{}, {}};
#pragma unroll
  for (int t = 0; t < 2; ++t) {
    v16bf bf;
#pragma unroll
    for (int s = 0; s < 16; ++s)
      bf[s] = wk1[(kbase + ((s < 8) ? s : (s + 8))) * 32 + t * 16 + (lane & 15)];
    acc[t] = WMMA_BF16(a, bf, acc[t]);
  }

#pragma unroll
  for (int t = 0; t < 2; ++t) {
#pragma unroll
    for (int r = 0; r < 8; ++r) {
      int M  = r + ((lane < 16) ? 0 : 8);
      size_t pr = (size_t)(wid * 16 + M);
      int co = t * 16 + (lane & 15);
      float v = acc[t][r] + b1[co];
      v = v > 0.0f ? v : 0.0f;
      act1[pr * 32 + co] = (bhalf)v;
    }
  }
}

// ---------------------------------------------------------------------------
// conv2: implicit GEMM, M = 32*64*64 pixels, N = 64, K = 288 (9 chunks of 32).
// The full 288x64 bf16 B panel (36,864 B) is DMA'd to LDS once per workgroup
// via the Tensor Data Mover (D# built per ISA ch.8: count=1, type=2,
// data_size=4B, one 9216-dword row), then waves read B fragments with
// ds_load. One wave = one pixel tile, all 4 N-tiles (A reused, 36 WMMAs).
// ---------------------------------------------------------------------------
__global__ void __launch_bounds__(256) conv2_kernel(const bhalf* __restrict__ act1,
                                                    const float* __restrict__ b2,
                                                    const bhalf* __restrict__ wk2,
                                                    bhalf* __restrict__ act2) {
  __shared__ __attribute__((aligned(128))) bhalf sw[18432];   // 288 x 64 bf16

  if (threadIdx.x < 32) {   // wave 0 issues the TDM op (EXEC ignored by TDM)
    unsigned lds = (unsigned)(uintptr_t)(&sw[0]);             // LDS byte offset
    unsigned long long ga = (unsigned long long)(uintptr_t)wk2;
    u32x4 g0;
    g0[0] = 1u;                                   // count=1 (valid user D#)
    g0[1] = lds;                                  // lds_addr
    g0[2] = (unsigned)(ga & 0xffffffffu);         // global_addr[31:0]
    g0[3] = (unsigned)((ga >> 32) & 0x01ffffffu)  // global_addr[56:32]
          | (2u << 30);                           // type = 2 ("image")
    i32x8 g1;
    g1[0] = 0x20000;                 // data_size=2 (4 bytes), no multicast
    g1[1] = (int)(9216u << 16);      // tensor_dim0[15:0] = 9216 dwords
    g1[2] = (int)(1u << 16);         // tensor_dim0[31:16]=0, tensor_dim1=1
    g1[3] = (int)(9216u << 16);      // tile_dim0 = 9216
    g1[4] = 1;                       // tile_dim1 = 1, tile_dim2 = 0
    g1[5] = 9216;                    // tensor_dim0_stride[31:0]
    g1[6] = 0;                       // stride0[47:32], stride1[15:0]
    g1[7] = 0;
    i32x4 g2 = {0, 0, 0, 0};
    i32x4 g3 = {0, 0, 0, 0};
    i32x8 g4 = {0, 0, 0, 0, 0, 0, 0, 0};          // 6-arg toolchain variant
    __builtin_amdgcn_tensor_load_to_lds(g0, g1, g2, g3, g4, 0);
    __builtin_amdgcn_s_wait_tensorcnt(0);         // s_wait_tensorcnt 0
  }
  __syncthreads();

  int wid  = blockIdx.x * (blockDim.x >> 5) + (threadIdx.x >> 5); // pixel tile
  int lane = threadIdx.x & 31;
  int row   = lane & 15;
  int kbase = (lane < 16) ? 0 : 8;

  int p   = wid * 16 + row;
  int b   = p >> 12;
  int rem = p & 4095;
  int oy  = rem >> 6, ox = rem & 63;

  v16bf zero16 = {};
  v8f acc[4] = {{}, {}, {}, {}};
#pragma unroll
  for (int k = 0; k < 9; ++k) {
    int ky = k / 3, kx = k % 3;
    int iy = 2 * oy + ky - 1;
    int ix = 2 * ox + kx - 1;
    union { v16bf v; v8bf h[2]; } a;
    if (iy >= 0 && iy < 128 && ix >= 0 && ix < 128) {
      const bhalf* src = act1 + (size_t)((b * 128 + iy) * 128 + ix) * 32;
      a.h[0] = *(const v8bf*)(src + kbase);
      a.h[1] = *(const v8bf*)(src + kbase + 16);
    } else {
      a.v = zero16;
    }
    const bhalf* brow = sw + (k * 32) * 64 + (lane & 15);   // LDS
#pragma unroll
    for (int t = 0; t < 4; ++t) {
      v16bf bf;
#pragma unroll
      for (int s = 0; s < 16; ++s)
        bf[s] = brow[(kbase + ((s < 8) ? s : (s + 8))) * 64 + t * 16];
      acc[t] = WMMA_BF16(a.v, bf, acc[t]);
    }
  }

#pragma unroll
  for (int t = 0; t < 4; ++t) {
#pragma unroll
    for (int r = 0; r < 8; ++r) {
      int M  = r + ((lane < 16) ? 0 : 8);
      size_t pr = (size_t)(wid * 16 + M);
      int co = t * 16 + (lane & 15);
      float v = acc[t][r] + b2[co];
      v = v > 0.0f ? v : 0.0f;
      act2[pr * 64 + co] = (bhalf)v;
    }
  }
}

// ---------------------------------------------------------------------------
// FC1: [32,262144] @ [262144,128], K split over 256 slabs of 1024.
// Block = 256 = 8 waves; wave = one 16-col N-tile; each w1 fragment (fp32,
// converted to bf16 in regs) feeds BOTH M-tiles -> 2 WMMAs per B load and
// w1 is read exactly once. f32 atomics combine the slabs.
// ---------------------------------------------------------------------------
__global__ void __launch_bounds__(256) fc1_kernel(const bhalf* __restrict__ act2,
                                                  const float* __restrict__ w1,
                                                  float* __restrict__ y) {
  int ntile = threadIdx.x >> 5;   // 0..7
  int lane  = threadIdx.x & 31;
  int slab  = blockIdx.x;         // 0..255, 1024 K each
  int row   = lane & 15;
  int kbase = (lane < 16) ? 0 : 8;

  const bhalf* a0row = act2 + (size_t)row * 262144;
  const bhalf* a1row = act2 + (size_t)(row + 16) * 262144;

  v8f c0 = {}, c1 = {};
  for (int kk = 0; kk < 32; ++kk) {
    int k0 = slab * 1024 + kk * 32;
    union { v16bf v; v8bf h[2]; } a0, a1;
    a0.h[0] = *(const v8bf*)(a0row + k0 + kbase);
    a0.h[1] = *(const v8bf*)(a0row + k0 + kbase + 16);
    a1.h[0] = *(const v8bf*)(a1row + k0 + kbase);
    a1.h[1] = *(const v8bf*)(a1row + k0 + kbase + 16);

    const float* wp = w1 + (size_t)k0 * 128 + ntile * 16 + (lane & 15);
    __builtin_prefetch(wp + 32 * 128, 0, 1);   // stream w1 (global_prefetch_b8)
    v16bf bf;
#pragma unroll
    for (int h = 0; h < 16; ++h) bf[h] = (bhalf)wp[(size_t)(kbase + ((h < 8) ? h : (h + 8))) * 128];

    c0 = WMMA_BF16(a0.v, bf, c0);
    c1 = WMMA_BF16(a1.v, bf, c1);
  }

#pragma unroll
  for (int r = 0; r < 8; ++r) {
    int M = r + ((lane < 16) ? 0 : 8);
    int N = ntile * 16 + (lane & 15);
    atomicAdd(&y[M * 128 + N], c0[r]);
    atomicAdd(&y[(M + 16) * 128 + N], c1[r]);
  }
}

// ---------------------------------------------------------------------------
// FC2: theta[b, i*3+j] = relu(y[b,:]+d1) @ w2 + d2   (32x6, one tiny block)
// ---------------------------------------------------------------------------
__global__ void fc2_kernel(const float* __restrict__ y, const float* __restrict__ d1,
                           const float* __restrict__ w2, const float* __restrict__ d2,
                           float* __restrict__ theta) {
  int t = threadIdx.x;
  if (t >= 192) return;
  int b = t / 6, j = t % 6;
  float s = d2[j];
  for (int k = 0; k < 128; ++k) {
    float v = y[b * 128 + k] + d1[k];
    v = v > 0.0f ? v : 0.0f;
    s += v * w2[k * 6 + j];
  }
  theta[b * 6 + j] = s;
}

// ---------------------------------------------------------------------------
// Bilinear grid sample with zero padding (GridSampler2D semantics).
// ---------------------------------------------------------------------------
__global__ void __launch_bounds__(256) sample_kernel(const float* __restrict__ in,
                                                     const float* __restrict__ theta,
                                                     float* __restrict__ out) {
  int id  = blockIdx.x * 256 + threadIdx.x;
  int b   = id >> 16;
  int rem = id & 65535;
  int yy  = rem >> 8, xx = rem & 255;

  const float* th = theta + b * 6;
  float X = (2.0f * xx + 1.0f) / 256.0f - 1.0f;
  float Y = (2.0f * yy + 1.0f) / 256.0f - 1.0f;
  float gx = th[0] * X + th[1] * Y + th[2];
  float gy = th[3] * X + th[4] * Y + th[5];
  float px = (gx + 1.0f) * 0.5f * 256.0f - 0.5f;
  float py = (gy + 1.0f) * 0.5f * 256.0f - 0.5f;
  float x0f = floorf(px), y0f = floorf(py);
  int   x0  = (int)x0f,   y0  = (int)y0f;
  float wx1 = px - x0f, wx0 = 1.0f - wx1;
  float wy1 = py - y0f, wy0 = 1.0f - wy1;

  float acc0 = 0.0f, acc1 = 0.0f, acc2 = 0.0f;
#pragma unroll
  for (int dy = 0; dy < 2; ++dy) {
    int yi = y0 + dy;
    float wy = dy ? wy1 : wy0;
    bool vy = (yi >= 0) && (yi < 256);
#pragma unroll
    for (int dx = 0; dx < 2; ++dx) {
      int xi = x0 + dx;
      float wx = dx ? wx1 : wx0;
      if (vy && xi >= 0 && xi < 256) {
        const float* s = in + ((size_t)(b * 256 + yi) * 256 + xi) * 3;
        float w = wy * wx;
        acc0 += s[0] * w; acc1 += s[1] * w; acc2 += s[2] * w;
      }
    }
  }
  float* o = out + (size_t)id * 3;
  o[0] = acc0; o[1] = acc1; o[2] = acc2;
}

// ---------------------------------------------------------------------------
extern "C" void kernel_launch(void* const* d_in, const int* in_sizes, int n_in,
                              void* d_out, int out_size, void* d_ws, size_t ws_size,
                              hipStream_t stream) {
  const float* in = (const float*)d_in[0];
  const float* k1 = (const float*)d_in[1];
  const float* b1 = (const float*)d_in[2];
  const float* k2 = (const float*)d_in[3];
  const float* b2 = (const float*)d_in[4];
  const float* w1 = (const float*)d_in[5];
  const float* d1 = (const float*)d_in[6];
  const float* w2 = (const float*)d_in[7];
  const float* d2 = (const float*)d_in[8];
  float* out = (float*)d_out;

  char* ws = (char*)d_ws;
  bhalf* wk1   = (bhalf*)(ws + 0);                      //   2 KB (32x32 bf16)
  bhalf* wk2   = (bhalf*)(ws + 4096);                   //  36 KB (288x64 bf16)
  float* y     = (float*)(ws + 40960);                  //  16 KB (32x128 f32)
  float* theta = (float*)(ws + 57344);                  //  768 B (32x6 f32)
  bhalf* act1  = (bhalf*)(ws + 65536);                  //  33.5 MB
  bhalf* act2  = (bhalf*)(ws + 65536 + 33554432ull);    //  16.8 MB

  pack_kernel<<<72, 256, 0, stream>>>(k1, k2, wk1, wk2, y);
  conv1_kernel<<<4096, 256, 0, stream>>>(in, b1, wk1, act1);   // 32768 waves
  conv2_kernel<<<1024, 256, 0, stream>>>(act1, b2, wk2, act2); //  8192 waves
  fc1_kernel<<<256, 256, 0, stream>>>(act2, w1, y);
  fc2_kernel<<<1, 192, 0, stream>>>(y, d1, w2, d2, theta);
  sample_kernel<<<8192, 256, 0, stream>>>(in, theta, out);
}
